// DeepONet_24936580121092
// MI455X (gfx1250) — matrix operand order repaired and checked by reading, not compiled, not run on hardware
//
#include <hip/hip_runtime.h>
#include <cmath>

// ---------------------------------------------------------------------------
// DeepONet on MI455X (gfx1250): all GEMMs via V_WMMA_F32_16X16X32_F16.
//  - weights pre-converted f32[K][N] -> f16[N][K] (NT layout: contiguous K)
//  - activations kept f16 row-major [M][K]
//  - final T@B^T reuses the same NT kernel (branch output is already [N][K])
//  - wave tile 32x64 (8 accumulators): 12 b128 loads per 8 WMMAs (1.5/WMMA)
// ---------------------------------------------------------------------------

typedef __attribute__((ext_vector_type(16))) _Float16 v16h;
typedef __attribute__((ext_vector_type(8)))  _Float16 v8h;
typedef __attribute__((ext_vector_type(8)))  float    v8f;

// f32 -> f16 elementwise (layout preserved)
__global__ void k_convert_f16(const float* __restrict__ in,
                              _Float16* __restrict__ out, int n) {
  int i = blockIdx.x * blockDim.x + threadIdx.x;
  if (i < n) out[i] = (_Float16)in[i];
}

// W f32 [K][N] -> Wt f16 [N][K]
__global__ void k_transpose_f16(const float* __restrict__ W,
                                _Float16* __restrict__ Wt, int K, int N) {
  int i = blockIdx.x * blockDim.x + threadIdx.x;
  if (i < K * N) {
    int n = i / K;
    int k = i - n * K;
    Wt[i] = (_Float16)W[(size_t)k * N + n];
  }
}

// trunk layer 1: K == 2, pure VALU. out = tanh(ys @ W + b), f16 row-major.
__global__ void k_trunk_l1(const float* __restrict__ ys,
                           const float* __restrict__ W /* [2][N] */,
                           const float* __restrict__ b,
                           _Float16* __restrict__ out, int M, int N) {
  int i = blockIdx.x * blockDim.x + threadIdx.x;
  if (i < M * N) {
    int m = i / N;
    int n = i - m * N;
    float v = fmaf(ys[2 * m + 1], W[N + n], fmaf(ys[2 * m], W[n], b[n]));
    out[i] = (_Float16)tanhf(v);
  }
}

// ---------------------------------------------------------------------------
// NT GEMM: out[M][N] = epilogue(A[M][K] @ Bt[N][K]^T + bias)
// Wave tile 32x64: 2 A fragments x 4 B fragments -> 8 WMMAs per K=32 step.
// 8 waves stacked along M -> block tile 256x64.
// Fragment layouts used here (A and B share contiguous K order):
//   A (16x32 f16): lane r=L%16 is row M=r; h=L/16 selects K-halves
//                  elem e -> K = e + 16*h (contig run of 16, loaded as two runs of 8)
//   B (32x16 f16): lane r is col N=r; elem e -> K = e + 16*h (contig run of 16)
//   C/D (16x16 f32): VGPR i -> row M = i + 8*h, col N = r
// ---------------------------------------------------------------------------
template <bool TANH, bool F32OUT, bool SCALAR_BIAS>
__global__ __launch_bounds__(256) void k_gemm_nt(
    const _Float16* __restrict__ A,   // [M][K] row-major f16
    const _Float16* __restrict__ Bt,  // [N][K] row-major f16
    const float* __restrict__ bias,   // [N], or [1] if SCALAR_BIAS
    void* __restrict__ outv,          // f16 [M][N] or f32 [M][N]
    int M, int N, int K) {
  const int lane = threadIdx.x & 31;
  const int wave = threadIdx.x >> 5;
  const int r = lane & 15;
  const int h = lane >> 4;

  const int mBase = blockIdx.y * 256 + wave * 32;
  const int nBase = blockIdx.x * 64;

  v8f acc[2][4];
#pragma unroll
  for (int m = 0; m < 2; ++m)
#pragma unroll
    for (int c = 0; c < 4; ++c) acc[m][c] = (v8f){};

  const _Float16* arow0 = A + (size_t)(mBase + r) * K;        // rows  0..15
  const _Float16* arow1 = A + (size_t)(mBase + 16 + r) * K;   // rows 16..31
  const _Float16* bp[4];
#pragma unroll
  for (int c = 0; c < 4; ++c)
    bp[c] = Bt + (size_t)(nBase + c * 16 + r) * K + 16 * h;

  for (int k = 0; k < K; k += 32) {
    // A fragments: K = {16h..16h+7} and {16h+8..16h+15}  (16B loads, aligned)
    v8h a0lo = *(const v8h*)(arow0 + k + 16 * h);
    v8h a0hi = *(const v8h*)(arow0 + k + 8 + 16 * h);
    v8h a1lo = *(const v8h*)(arow1 + k + 16 * h);
    v8h a1hi = *(const v8h*)(arow1 + k + 8 + 16 * h);
    v16h a0, a1;
#pragma unroll
    for (int i = 0; i < 8; ++i) {
      a0[i] = a0lo[i]; a0[i + 8] = a0hi[i];
      a1[i] = a1lo[i]; a1[i + 8] = a1hi[i];
    }

    // B fragments: K = {16h..16h+15} at fixed column (32B loads, aligned)
    v16h b0 = *(const v16h*)(bp[0] + k);
    v16h b1 = *(const v16h*)(bp[1] + k);
    v16h b2 = *(const v16h*)(bp[2] + k);
    v16h b3 = *(const v16h*)(bp[3] + k);

    acc[0][0] = __builtin_amdgcn_wmma_f32_16x16x32_f16(false, a0, false, b0, (short)0, acc[0][0], false, false);
    acc[0][1] = __builtin_amdgcn_wmma_f32_16x16x32_f16(false, a0, false, b1, (short)0, acc[0][1], false, false);
    acc[0][2] = __builtin_amdgcn_wmma_f32_16x16x32_f16(false, a0, false, b2, (short)0, acc[0][2], false, false);
    acc[0][3] = __builtin_amdgcn_wmma_f32_16x16x32_f16(false, a0, false, b3, (short)0, acc[0][3], false, false);
    acc[1][0] = __builtin_amdgcn_wmma_f32_16x16x32_f16(false, a1, false, b0, (short)0, acc[1][0], false, false);
    acc[1][1] = __builtin_amdgcn_wmma_f32_16x16x32_f16(false, a1, false, b1, (short)0, acc[1][1], false, false);
    acc[1][2] = __builtin_amdgcn_wmma_f32_16x16x32_f16(false, a1, false, b2, (short)0, acc[1][2], false, false);
    acc[1][3] = __builtin_amdgcn_wmma_f32_16x16x32_f16(false, a1, false, b3, (short)0, acc[1][3], false, false);
  }

#pragma unroll
  for (int m = 0; m < 2; ++m) {
#pragma unroll
    for (int c = 0; c < 4; ++c) {
      const v8f av = acc[m][c];
      const int col = nBase + c * 16 + r;
      const float bv = SCALAR_BIAS ? bias[0] : bias[col];
      if (F32OUT) {
        float* out = (float*)outv;
#pragma unroll
        for (int i = 0; i < 8; ++i) {
          int row = mBase + 16 * m + i + 8 * h;
          out[(size_t)row * N + col] = av[i] + bv;
        }
      } else {
        _Float16* out = (_Float16*)outv;
#pragma unroll
        for (int i = 0; i < 8; ++i) {
          int row = mBase + 16 * m + i + 8 * h;
          float v = av[i] + bv;
          if (TANH) v = tanhf(v);
          out[(size_t)row * N + col] = (_Float16)v;
        }
      }
    }
  }
}

// ---------------------------------------------------------------------------

extern "C" void kernel_launch(void* const* d_in, const int* in_sizes, int n_in,
                              void* d_out, int out_size, void* d_ws, size_t ws_size,
                              hipStream_t stream) {
  (void)in_sizes; (void)n_in; (void)out_size; (void)ws_size;

  const float* u_in = (const float*)d_in[0];           // [2048, 256]
  const float* ys   = (const float*)d_in[1];           // [16384, 2]
  const float* bw[4] = {(const float*)d_in[2], (const float*)d_in[3],
                        (const float*)d_in[4], (const float*)d_in[5]};
  const float* bb[4] = {(const float*)d_in[6], (const float*)d_in[7],
                        (const float*)d_in[8], (const float*)d_in[9]};
  const float* tw[4] = {(const float*)d_in[10], (const float*)d_in[11],
                        (const float*)d_in[12], (const float*)d_in[13]};
  const float* tb[4] = {(const float*)d_in[14], (const float*)d_in[15],
                        (const float*)d_in[16], (const float*)d_in[17]};
  const float* b0 = (const float*)d_in[18];            // scalar

  // --- workspace layout (256B aligned), ~42 MB total ---
  char* ws = (char*)d_ws;
  size_t off = 0;
  auto alloc_h = [&](size_t nelem) -> _Float16* {
    off = (off + 255) & ~(size_t)255;
    _Float16* p = (_Float16*)(ws + off);
    off += nelem * sizeof(_Float16);
    return p;
  };
  _Float16* bwT0 = alloc_h(512 * 256);    // [512][256]
  _Float16* bwT1 = alloc_h(512 * 512);
  _Float16* bwT2 = alloc_h(512 * 512);
  _Float16* bwT3 = alloc_h(256 * 512);    // [256][512]
  _Float16* twT1 = alloc_h(512 * 512);
  _Float16* twT2 = alloc_h(512 * 512);
  _Float16* twT3 = alloc_h(256 * 512);
  _Float16* u16  = alloc_h(2048 * 256);
  _Float16* aB0  = alloc_h(2048 * 512);
  _Float16* aB1  = alloc_h(2048 * 512);
  _Float16* aT0  = alloc_h((size_t)16384 * 512);
  _Float16* aT1  = alloc_h((size_t)16384 * 512);

  const int T = 256;
  auto g1 = [](int n, int t) { return (n + t - 1) / t; };

  // --- stage 0: weight transpose+convert, input convert ---
  k_transpose_f16<<<g1(256 * 512, T), T, 0, stream>>>(bw[0], bwT0, 256, 512);
  k_transpose_f16<<<g1(512 * 512, T), T, 0, stream>>>(bw[1], bwT1, 512, 512);
  k_transpose_f16<<<g1(512 * 512, T), T, 0, stream>>>(bw[2], bwT2, 512, 512);
  k_transpose_f16<<<g1(512 * 256, T), T, 0, stream>>>(bw[3], bwT3, 512, 256);
  k_transpose_f16<<<g1(512 * 512, T), T, 0, stream>>>(tw[1], twT1, 512, 512);
  k_transpose_f16<<<g1(512 * 512, T), T, 0, stream>>>(tw[2], twT2, 512, 512);
  k_transpose_f16<<<g1(512 * 256, T), T, 0, stream>>>(tw[3], twT3, 512, 256);
  k_convert_f16<<<g1(2048 * 256, T), T, 0, stream>>>(u_in, u16, 2048 * 256);

  const dim3 blk(256);
  // --- branch MLP: [2048] rows, block tile 256x64 ---
  k_gemm_nt<true,  false, false><<<dim3(512 / 64, 2048 / 256), blk, 0, stream>>>(
      u16, bwT0, bb[0], aB0, 2048, 512, 256);
  k_gemm_nt<true,  false, false><<<dim3(512 / 64, 2048 / 256), blk, 0, stream>>>(
      aB0, bwT1, bb[1], aB1, 2048, 512, 512);
  k_gemm_nt<true,  false, false><<<dim3(512 / 64, 2048 / 256), blk, 0, stream>>>(
      aB1, bwT2, bb[2], aB0, 2048, 512, 512);
  k_gemm_nt<false, false, false><<<dim3(256 / 64, 2048 / 256), blk, 0, stream>>>(
      aB0, bwT3, bb[3], aB1, 2048, 256, 512);          // B = aB1 [2048][256]

  // --- trunk MLP: [16384] rows ---
  k_trunk_l1<<<g1(16384 * 512, T), T, 0, stream>>>(ys, tw[0], tb[0], aT0, 16384, 512);
  k_gemm_nt<true,  false, false><<<dim3(512 / 64, 16384 / 256), blk, 0, stream>>>(
      aT0, twT1, tb[1], aT1, 16384, 512, 512);
  k_gemm_nt<true,  false, false><<<dim3(512 / 64, 16384 / 256), blk, 0, stream>>>(
      aT1, twT2, tb[2], aT0, 16384, 512, 512);
  k_gemm_nt<false, false, false><<<dim3(256 / 64, 16384 / 256), blk, 0, stream>>>(
      aT0, twT3, tb[3], aT1, 16384, 256, 512);         // T = aT1 [16384][256]

  // --- final: out = T @ B^T + b0  ->  f32 [16384][2048] ---
  k_gemm_nt<false, true, true><<<dim3(2048 / 64, 16384 / 256), blk, 0, stream>>>(
      aT1, aB1, b0, d_out, 16384, 2048, 256);
}
